// FeatureInteractionNEtowrk_32255204393218
// MI455X (gfx1250) — matrix operand order repaired and checked by reading, not compiled, not run on hardware
//
#include <hip/hip_runtime.h>
#include <math.h>

typedef __attribute__((ext_vector_type(16))) _Float16 v16h;
typedef __attribute__((ext_vector_type(8)))  _Float16 v8h;
typedef __attribute__((ext_vector_type(8)))  float    v8f;

namespace fin {

constexpr int BSZ = 8;
constexpr int C   = 256;
constexpr int L   = 1600;      // 40*40 tokens
constexpr int M   = BSZ * L;   // 12800 rows
constexpr int HD  = 32;
constexpr int NH  = 8;

// ---------------- WMMA fragment loads (gfx1250, wave32) ----------------
// A-matrix 16x32 f16 (ISA 7.12.2): lane row = lane&15; lanes>=16 carry K+8;
// halves 0..7 -> K = h, halves 8..15 -> K = h+8+8. Two contiguous 16B loads.
__device__ __forceinline__ v16h load_frag_a(const _Float16* __restrict__ base,
                                            int ld, int row0, int k0) {
  const int lane = threadIdx.x & 31;
  const _Float16* p = base + (size_t)(row0 + (lane & 15)) * ld
                           + (k0 + ((lane >> 4) << 3));
  v8h lo = *(const v8h*)(p);
  v8h hi = *(const v8h*)(p + 16);
  v16h f;
#pragma unroll
  for (int i = 0; i < 8; ++i) { f[i] = lo[i]; f[i + 8] = hi[i]; }
  return f;
}

// B-matrix 32x16 f16: lane column = lane&15; same K order as A: lanes>=16
// carry K+8; halves 0..7 -> K = h, halves 8..15 -> K = h+8+8. Source stored
// (N,K) row-major (i.e. W for X@W^T), so two contiguous 16B loads per lane.
__device__ __forceinline__ v16h load_frag_b(const _Float16* __restrict__ baseNK,
                                            int ldk, int n0, int k0) {
  const int lane = threadIdx.x & 31;
  const _Float16* p = baseNK + (size_t)(n0 + (lane & 15)) * ldk
                             + (k0 + ((lane >> 4) << 3));
  v8h lo = *(const v8h*)(p);
  v8h hi = *(const v8h*)(p + 16);
  v16h f;
#pragma unroll
  for (int i = 0; i < 8; ++i) { f[i] = lo[i]; f[i + 8] = hi[i]; }
  return f;
}

__device__ __forceinline__ v8f wmma_f16(v16h a, v16h b, v8f c) {
  return __builtin_amdgcn_wmma_f32_16x16x32_f16(false, a, false, b,
                                                (short)0, c, false, false);
}

// sin/cos positional embedding value for token l, channel c (dim=256, pd=64)
__device__ __forceinline__ float pos_val(int l, int c) {
  const int r  = l / 40, cl = l - r * 40;
  const int seg = c >> 6;
  const int j   = c & 63;
  const float omega = __expf(-(float)j * (9.210340371976184f / 64.0f)); // 10000^{-j/64}
  const float t = (seg < 2 ? (float)r : (float)cl) * omega;
  return (seg & 1) ? __cosf(t) : __sinf(t);
}

// ---------------- kernels ----------------

__global__ void __launch_bounds__(256)
k_f32_to_f16(const float* __restrict__ in, _Float16* __restrict__ out, int n) {
  const int i = blockIdx.x * 256 + threadIdx.x;
  if (i < n) out[i] = (_Float16)in[i];
}

// low (8,256,160,160) -> 4x4 avg pool, +pos -> Xlow f16 (M,256); resid f32 (pool only)
__global__ void __launch_bounds__(256)
k_pool_low(const float* __restrict__ low, _Float16* __restrict__ X,
           float* __restrict__ resid) {
  const int tid = blockIdx.x * 256 + threadIdx.x;   // 8*256*40*40 threads
  const int cl = tid % 40;
  int t = tid / 40;
  const int r = t % 40; t /= 40;
  const int c = t % 256;
  const int b = t / 256;
  const float4* row = (const float4*)low
                    + ((size_t)((b * 256 + c) * 160 + 4 * r)) * 40 + cl;
  float s = 0.f;
#pragma unroll
  for (int i = 0; i < 4; ++i) { float4 v = row[(size_t)i * 40]; s += v.x + v.y + v.z + v.w; }
  const float mean = s * (1.0f / 16.0f);
  const int l = r * 40 + cl;
  const size_t o = (size_t)(b * L + l) * C + c;
  resid[o] = mean;
  X[o] = (_Float16)(mean + pos_val(l, c));
}

// high (8,256,80,80) -> 2x2 avg pool, +pos -> Xhigh f16 (M,256)
__global__ void __launch_bounds__(256)
k_pool_high(const float* __restrict__ high, _Float16* __restrict__ X) {
  const int tid = blockIdx.x * 256 + threadIdx.x;
  const int cl = tid % 40;
  int t = tid / 40;
  const int r = t % 40; t /= 40;
  const int c = t % 256;
  const int b = t / 256;
  const float2* row = (const float2*)high
                    + ((size_t)((b * 256 + c) * 80 + 2 * r)) * 40 + cl;
  float s = 0.f;
#pragma unroll
  for (int i = 0; i < 2; ++i) { float2 v = row[(size_t)i * 40]; s += v.x + v.y; }
  const float mean = s * 0.25f;
  const int l = r * 40 + cl;
  const size_t o = (size_t)(b * L + l) * C + c;
  X[o] = (_Float16)(mean + pos_val(l, c));
}

// Generic WMMA GEMM: out = epi(A(M,K) @ W(N,K)^T + bias)
// block = 256 thr = 8 waves; wave tile 16x64; block tile 32(M) x 256(N)
// EPI: 0 = store f16, 1 = exact GELU -> f16, 2 = store f32
template <int EPI>
__global__ void __launch_bounds__(256)
k_gemm(const _Float16* __restrict__ A, const _Float16* __restrict__ W,
       const float* __restrict__ bias, void* __restrict__ out, int N, int K) {
  const int wave = threadIdx.x >> 5;
  const int row0 = blockIdx.y * 32  + (wave >> 2) * 16;
  const int col0 = blockIdx.x * 256 + (wave & 3) * 64;
  v8f acc[4] = {};
  for (int k0 = 0; k0 < K; k0 += 32) {
    if (k0 + 32 < K) __builtin_prefetch(A + (size_t)row0 * K + k0 + 32, 0, 3);
    v16h a = load_frag_a(A, K, row0, k0);
#pragma unroll
    for (int j = 0; j < 4; ++j) {
      v16h b = load_frag_b(W, K, col0 + j * 16, k0);
      acc[j] = wmma_f16(a, b, acc[j]);
    }
  }
  const int lane  = threadIdx.x & 31;
  const int nlane = lane & 15;
  const int mofs  = (lane >> 4) << 3;   // C/D layout: lanes>=16 carry M+8
#pragma unroll
  for (int j = 0; j < 4; ++j) {
    const int n = col0 + j * 16 + nlane;
    const float bv = bias[n];
#pragma unroll
    for (int r = 0; r < 8; ++r) {
      const size_t o = (size_t)(row0 + mofs + r) * N + n;
      float v = acc[j][r] + bv;
      if (EPI == 1) v = 0.5f * v * (1.0f + erff(v * 0.70710678118654752f));
      if (EPI == 2) ((float*)out)[o] = v;
      else          ((_Float16*)out)[o] = (_Float16)v;
    }
  }
}

// Per-(head,dim) scales: 1/max(sum_l|Q|,eps) * 1/max(sum_l|K|,eps). One block per head.
__global__ void __launch_bounds__(256)
k_colsum_scale(const _Float16* __restrict__ Q, const _Float16* __restrict__ Kt,
               float* __restrict__ scale) {
  __shared__ float sQ[256], sK[256];
  const int head = blockIdx.x;            // b*8+h
  const int b = head >> 3, h = head & 7;
  const int t = threadIdx.x;
  const int d = t & 31, slice = t >> 5;   // 8 row slices
  float aq = 0.f, ak = 0.f;
  for (int l = slice; l < L; l += 8) {
    const size_t idx = (size_t)(b * L + l) * C + h * HD + d;
    aq += fabsf((float)Q[idx]);
    ak += fabsf((float)Kt[idx]);
  }
  sQ[t] = aq; sK[t] = ak;
  __syncthreads();
  if (t < 32) {
    float q = 0.f, k = 0.f;
#pragma unroll
    for (int s = 0; s < 8; ++s) { q += sQ[s * 32 + t]; k += sK[s * 32 + t]; }
    scale[head * HD + t] = (1.0f / fmaxf(q, 1e-6f)) * (1.0f / fmaxf(k, 1e-6f));
  }
}

// KVt[head][e][d] = scale[head][d] * sum_l K[l][d] * V[l][e]   (32x32 per head)
__global__ void __launch_bounds__(256)
k_kv(const _Float16* __restrict__ Kt, const _Float16* __restrict__ V,
     const float* __restrict__ scale, _Float16* __restrict__ KVt) {
  __shared__ float lk[32][33];
  __shared__ float lv[32][33];
  const int head = blockIdx.x;
  const int b = head >> 3, h = head & 7;
  const int t = threadIdx.x;
  const int d  = t >> 3;          // 0..31
  const int e0 = (t & 7) * 4;     // 4 outputs per thread
  float acc[4] = {0.f, 0.f, 0.f, 0.f};
  for (int l0 = 0; l0 < L; l0 += 32) {
#pragma unroll
    for (int i = 0; i < 4; ++i) {
      const int idx = t + i * 256;        // 0..1023
      const int ll = idx >> 5, cc = idx & 31;
      const size_t g = (size_t)(b * L + l0 + ll) * C + h * HD + cc;
      lk[ll][cc] = (float)Kt[g];
      lv[ll][cc] = (float)V[g];
    }
    __syncthreads();
#pragma unroll 8
    for (int ll = 0; ll < 32; ++ll) {
      const float kd = lk[ll][d];
#pragma unroll
      for (int j = 0; j < 4; ++j) acc[j] += kd * lv[ll][e0 + j];
    }
    __syncthreads();
  }
  const float sc = scale[head * HD + d];
#pragma unroll
  for (int j = 0; j < 4; ++j)
    KVt[(size_t)head * (HD * HD) + (e0 + j) * HD + d] = (_Float16)(acc[j] * sc);
}

// attn = Q_head(16x32 tile) @ KVt^T (32x32) via two WMMAs; one wave per block
__global__ void __launch_bounds__(32)
k_attn(const _Float16* __restrict__ Q, const _Float16* __restrict__ KVt,
       float* __restrict__ attn) {
  const int head = blockIdx.y;
  const int b = head >> 3, h = head & 7;
  const int l0 = blockIdx.x * 16;
  const _Float16* qb = Q + (size_t)(b * L + l0) * C + h * HD;
  const _Float16* kv = KVt + (size_t)head * (HD * HD);  // (e,d) row-major = B(N,K)
  const v16h a = load_frag_a(qb, C, 0, 0);
  v8f acc0 = {}, acc1 = {};
  acc0 = wmma_f16(a, load_frag_b(kv, HD, 0, 0),  acc0);
  acc1 = wmma_f16(a, load_frag_b(kv, HD, 16, 0), acc1);
  const int lane = threadIdx.x & 31;
  const int nlane = lane & 15;
  const int mofs = (lane >> 4) << 3;
#pragma unroll
  for (int r = 0; r < 8; ++r) {
    const size_t row = (size_t)(b * L + l0 + mofs + r) * C + h * HD;
    attn[row + nlane]      = acc0[r];
    attn[row + 16 + nlane] = acc1[r];
  }
}

// RMS-norm over each 32-wide head group; one thread per group
__global__ void __launch_bounds__(256)
k_rmsnorm(const float* __restrict__ attn, const float* __restrict__ rms_w,
          _Float16* __restrict__ out) {
  const int g = blockIdx.x * 256 + threadIdx.x;   // M*NH groups
  const float* p = attn + (size_t)g * HD;
  float ss = 0.f;
#pragma unroll
  for (int i = 0; i < HD; ++i) { const float v = p[i]; ss += v * v; }
  const float sc = rsqrtf(ss * (1.0f / HD) + 1.1920929e-07f);
  _Float16* o = out + (size_t)g * HD;
#pragma unroll
  for (int i = 0; i < HD; ++i) o[i] = (_Float16)(p[i] * sc * rms_w[i]);
}

// y = resid + LayerNorm(X)*w+b ; wave per row (256 elems), shuffle reductions
__global__ void __launch_bounds__(256)
k_ln_residual(const float* __restrict__ X, const float* __restrict__ resid,
              const float* __restrict__ w, const float* __restrict__ bb,
              float* __restrict__ of32, _Float16* __restrict__ of16) {
  const int row  = blockIdx.x * 8 + (threadIdx.x >> 5);
  const int lane = threadIdx.x & 31;
  const float* p = X + (size_t)row * C;
  float v[8], s = 0.f;
#pragma unroll
  for (int i = 0; i < 8; ++i) { v[i] = p[lane + 32 * i]; s += v[i]; }
#pragma unroll
  for (int o = 16; o > 0; o >>= 1) s += __shfl_xor(s, o, 32);
  const float mean = s * (1.0f / C);
  float vs = 0.f;
#pragma unroll
  for (int i = 0; i < 8; ++i) { const float d = v[i] - mean; vs += d * d; }
#pragma unroll
  for (int o = 16; o > 0; o >>= 1) vs += __shfl_xor(vs, o, 32);
  const float inv = rsqrtf(vs * (1.0f / C) + 1e-5f);
  const float* rp = resid + (size_t)row * C;
#pragma unroll
  for (int i = 0; i < 8; ++i) {
    const int cc = lane + 32 * i;
    const float y = (v[i] - mean) * inv * w[cc] + bb[cc] + rp[cc];
    if (of32) of32[(size_t)row * C + cc] = y;
    if (of16) of16[(size_t)row * C + cc] = (_Float16)y;
  }
}

// bilinear 40->160 (half-pixel centers, edge clamp) fused with *low
__global__ void __launch_bounds__(256)
k_upsample_mul(const float* __restrict__ f40, const float* __restrict__ low,
               float* __restrict__ out) {
  const size_t tid = (size_t)blockIdx.x * 256 + threadIdx.x;  // 8*256*160*160
  const int x = (int)(tid % 160);
  size_t t = tid / 160;
  const int y = (int)(t % 160); t /= 160;
  const int c = (int)(t % 256);
  const int b = (int)(t / 256);
  const float sx = x * 0.25f - 0.375f;
  const float sy = y * 0.25f - 0.375f;
  const float fx = floorf(sx), fy = floorf(sy);
  const float wx = sx - fx,    wy = sy - fy;
  const int x0 = min(39, max(0, (int)fx));
  const int x1 = min(39, max(0, (int)fx + 1));
  const int y0 = min(39, max(0, (int)fy));
  const int y1 = min(39, max(0, (int)fy + 1));
  auto at = [&](int yy, int xx) -> float {
    return f40[((size_t)b * L + yy * 40 + xx) * C + c];
  };
  const float v0 = at(y0, x0) * (1.f - wx) + at(y0, x1) * wx;
  const float v1 = at(y1, x0) * (1.f - wx) + at(y1, x1) * wx;
  out[tid] = (v0 * (1.f - wy) + v1 * wy) * low[tid];
}

}  // namespace fin

extern "C" void kernel_launch(void* const* d_in, const int* in_sizes, int n_in,
                              void* d_out, int out_size, void* d_ws, size_t ws_size,
                              hipStream_t stream) {
  using namespace fin;
  (void)in_sizes; (void)n_in; (void)out_size;

  const float* low   = (const float*)d_in[0];
  const float* high  = (const float*)d_in[1];
  const float* q_w   = (const float*)d_in[2];
  const float* q_b   = (const float*)d_in[3];
  const float* k_w   = (const float*)d_in[4];
  const float* k_b   = (const float*)d_in[5];
  const float* v_w   = (const float*)d_in[6];
  const float* v_b   = (const float*)d_in[7];
  const float* o_w   = (const float*)d_in[8];
  const float* o_b   = (const float*)d_in[9];
  const float* rms_w = (const float*)d_in[10];
  const float* l1_w  = (const float*)d_in[11];
  const float* l1_b  = (const float*)d_in[12];
  const float* l2_w  = (const float*)d_in[13];
  const float* l2_b  = (const float*)d_in[14];
  const float* n1_w  = (const float*)d_in[15];
  const float* n1_b  = (const float*)d_in[16];
  const float* n2_w  = (const float*)d_in[17];
  const float* n2_b  = (const float*)d_in[18];
  float* out = (float*)d_out;

  char* wsp = (char*)d_ws;
  size_t off = 0;
  auto ALLOC = [&](size_t bytes) -> void* {
    void* p = wsp + off;
    off += (bytes + 255) & ~(size_t)255;
    return p;
  };
  (void)ws_size;

  const size_t MC = (size_t)M * C;
  _Float16* Xlow   = (_Float16*)ALLOC(MC * 2);        // reused later as attn_n
  _Float16* Xhigh  = (_Float16*)ALLOC(MC * 2);        // reused later as out1 f16
  float*    resid  = (float*)   ALLOC(MC * 4);        // reused later as final40
  _Float16* Qh     = (_Float16*)ALLOC(MC * 2);
  _Float16* Kh     = (_Float16*)ALLOC(MC * 2);
  _Float16* Vh     = (_Float16*)ALLOC(MC * 2);
  float*    scale  = (float*)   ALLOC(64 * HD * 4);
  _Float16* KVt    = (_Float16*)ALLOC(64 * HD * HD * 2);
  float*    big1   = (float*)   ALLOC(MC * 4);        // attn -> O -> M2 (sequential reuse)
  float*    resid2 = (float*)   ALLOC(MC * 4);
  _Float16* Hmid   = (_Float16*)ALLOC((size_t)M * 1024 * 2);
  _Float16* qw16   = (_Float16*)ALLOC((size_t)C * C * 2);
  _Float16* kw16   = (_Float16*)ALLOC((size_t)C * C * 2);
  _Float16* vw16   = (_Float16*)ALLOC((size_t)C * C * 2);
  _Float16* ow16   = (_Float16*)ALLOC((size_t)C * C * 2);
  _Float16* l1w16  = (_Float16*)ALLOC((size_t)4 * C * C * 2);
  _Float16* l2w16  = (_Float16*)ALLOC((size_t)4 * C * C * 2);

  _Float16* attn_n = Xlow;    // Xlow consumed by Q-GEMM before this is written
  _Float16* out1h  = Xhigh;   // Xhigh consumed by K/V-GEMMs before this is written
  float*    final40 = resid;  // resid consumed by first ln_residual

  // 1) weight conversion f32 -> f16
  k_f32_to_f16<<<(C * C + 255) / 256, 256, 0, stream>>>(q_w, qw16, C * C);
  k_f32_to_f16<<<(C * C + 255) / 256, 256, 0, stream>>>(k_w, kw16, C * C);
  k_f32_to_f16<<<(C * C + 255) / 256, 256, 0, stream>>>(v_w, vw16, C * C);
  k_f32_to_f16<<<(C * C + 255) / 256, 256, 0, stream>>>(o_w, ow16, C * C);
  k_f32_to_f16<<<(4 * C * C + 255) / 256, 256, 0, stream>>>(l1_w, l1w16, 4 * C * C);
  k_f32_to_f16<<<(4 * C * C + 255) / 256, 256, 0, stream>>>(l2_w, l2w16, 4 * C * C);

  // 2) pooling + positional embedding (memory-bound main read of `low`)
  k_pool_low <<<(M * C) / 256, 256, 0, stream>>>(low, Xlow, resid);
  k_pool_high<<<(M * C) / 256, 256, 0, stream>>>(high, Xhigh);

  // 3) Q/K/V projections (WMMA f16, f32 accumulate)
  dim3 g256(1, M / 32);
  k_gemm<0><<<g256, 256, 0, stream>>>(Xlow,  qw16, q_b, Qh, C, C);
  k_gemm<0><<<g256, 256, 0, stream>>>(Xhigh, kw16, k_b, Kh, C, C);
  k_gemm<0><<<g256, 256, 0, stream>>>(Xhigh, vw16, v_b, Vh, C, C);

  // 4) L1-norm column scales folded into KV
  k_colsum_scale<<<64, 256, 0, stream>>>(Qh, Kh, scale);
  k_kv<<<64, 256, 0, stream>>>(Kh, Vh, scale, KVt);

  // 5) attn = Q @ KV' (WMMA per head), then RMS-norm
  k_attn<<<dim3(L / 16, 64), 32, 0, stream>>>(Qh, KVt, big1);
  k_rmsnorm<<<(M * NH) / 256, 256, 0, stream>>>(big1, rms_w, attn_n);

  // 6) output projection + LN + residual
  k_gemm<2><<<g256, 256, 0, stream>>>(attn_n, ow16, o_b, big1, C, C);
  k_ln_residual<<<M / 8, 256, 0, stream>>>(big1, resid, n1_w, n1_b, resid2, out1h);

  // 7) MLP: 256 -> 1024 (GELU) -> 256, + LN + residual
  k_gemm<1><<<dim3(4, M / 32), 256, 0, stream>>>(out1h, l1w16, l1_b, Hmid, 1024, C);
  k_gemm<2><<<g256, 256, 0, stream>>>(Hmid, l2w16, l2_b, big1, C, 1024);
  k_ln_residual<<<M / 8, 256, 0, stream>>>(big1, resid2, n2_w, n2_b, final40,
                                           (_Float16*)nullptr);

  // 8) bilinear 4x upsample fused with elementwise multiply by original low
  k_upsample_mul<<<(8UL * 256 * 160 * 160) / 256, 256, 0, stream>>>(final40, low, out);
}
